// SpMiddlePillarEncoder8x18_61478161875111
// MI455X (gfx1250) — matrix operand order repaired and checked by reading, not compiled, not run on hardware
//
#include <hip/hip_runtime.h>
#include <hip/hip_bf16.h>
#include <stdint.h>

// ---------------------------------------------------------------------------
// SpMiddlePillarEncoder for MI455X (gfx1250, wave32, WMMA bf16)
//   - convs via implicit GEMM with v_wmma_f32_16x16x32_bf16
//   - one wave = 16 output pixels x 256 output channels (128 acc VGPRs)
//   - weights pre-packed into per-lane WMMA B-fragment order
//   - B fragments double-buffered in LDS via global_load_async_to_lds_b128
//     (ASYNCcnt); inside a k-step the B fragment rotates through registers
//     with distance-2 prefetch so ds_load latency overlaps WMMA execution
// ---------------------------------------------------------------------------

#define CH     256
#define GRIDN  188
#define GRIDN5 94
#define NB     2
#define PC0    (-75.2f)
#define BEVINV 1.25f   // 1/0.8

#define KSTEPS     72          // 9 taps * 8 k-chunks of 32 channels
#define STEP_BYTES 16384       // 16 frags * 32 lanes * 16 halfs * 2B

typedef __attribute__((ext_vector_type(16))) __bf16        v16bf;
typedef __attribute__((ext_vector_type(8)))  float         v8f;
typedef __attribute__((ext_vector_type(8)))  unsigned short v8u;
typedef __attribute__((ext_vector_type(16))) unsigned short v16u;

__device__ __forceinline__ unsigned short f2b(float f) {
  unsigned u = __float_as_uint(f);
  u += 0x7FFFu + ((u >> 16) & 1u);          // round-to-nearest-even
  return (unsigned short)(u >> 16);
}
__device__ __forceinline__ float b2f(unsigned short h) {
  return __uint_as_float(((unsigned)h) << 16);
}

// ---------------------------------------------------------------------------
// Elementwise helpers
// ---------------------------------------------------------------------------
__global__ void zero_f32(float* __restrict__ p, size_t n) {
  size_t i = (size_t)blockIdx.x * blockDim.x + threadIdx.x;
  if (i < n) p[i] = 0.0f;
}

__global__ void cvt_f32_bf16(const float* __restrict__ s, unsigned short* __restrict__ d, size_t n) {
  size_t i = (size_t)blockIdx.x * blockDim.x + threadIdx.x;
  if (i < n) d[i] = f2b(s[i]);
}

// bf16 NHWC -> f32 NCHW
__global__ void nhwc_to_nchw(const unsigned short* __restrict__ s, float* __restrict__ d,
                             int H, int W) {
  size_t n = (size_t)NB * CH * H * W;
  size_t i = (size_t)blockIdx.x * blockDim.x + threadIdx.x;
  if (i >= n) return;
  int x = (int)(i % W); size_t t = i / W;
  int y = (int)(t % H); t /= H;
  int c = (int)(t % CH);
  int b = (int)(t / CH);
  d[i] = b2f(s[((size_t)(b * H + y) * W + x) * CH + c]);
}

// ---------------------------------------------------------------------------
// Pack one conv weight tensor w[3][3][256][256] (ky,kx,ci,co fp32) into
// WMMA B-fragment order: frag f = (tap*8+kc)*16+nt, 32 lanes x 16 halfs each.
// Lane L<16: col=L, K=e (0..15); lane>=16: col=L-16, K=16+e.  ci=kc*32+K.
// ---------------------------------------------------------------------------
__global__ void pack_weights(const float* __restrict__ w, unsigned short* __restrict__ wp) {
  int t = blockIdx.x * blockDim.x + threadIdx.x;
  if (t >= 9 * 8 * 16 * 32) return;
  int lane = t & 31;
  int f    = t >> 5;
  int nt   = f & 15;
  int kc   = (f >> 4) & 7;
  int tap  = f >> 7;
  int hi   = lane >> 4;
  int col  = lane & 15;
  int co   = nt * 16 + col;
  int cib  = kc * 32 + hi * 16;
  unsigned short* dst = wp + (size_t)f * 512 + lane * 16;
  const float* src = w + (size_t)tap * CH * CH + co;
#pragma unroll
  for (int e = 0; e < 16; ++e)
    dst[e] = f2b(src[(size_t)(cib + e) * CH]);
}

// ---------------------------------------------------------------------------
// Point featurization + MLP(8->256) + BN + ReLU + scatter-max into grid.
// Post-ReLU values are >= 0 so int-bitpattern atomicMax on a zeroed grid is
// exactly the reference's  where(occ>0, max(h), 0).
// ---------------------------------------------------------------------------
__global__ __launch_bounds__(256) void points_kernel(
    const float* __restrict__ xyz, const float* __restrict__ ptf,
    const int* __restrict__ bcnt,  const float* __restrict__ w,
    const float* __restrict__ g,   const float* __restrict__ bi,
    float* __restrict__ grid,      float* __restrict__ maskp, int n) {
  __shared__ float sw[8 * CH];
  for (int i = threadIdx.x; i < 8 * CH; i += blockDim.x) sw[i] = w[i];
  __syncthreads();
  int idx = blockIdx.x * blockDim.x + threadIdx.x;
  if (idx >= n) return;
  int b = (idx < bcnt[0]) ? 0 : 1;
  float x = xyz[idx * 3 + 0], y = xyz[idx * 3 + 1], zc = xyz[idx * 3 + 2];
  int ix = (int)floorf((x - PC0) * BEVINV);
  int iy = (int)floorf((y - PC0) * BEVINV);
  ix = ix < 0 ? 0 : (ix > GRIDN - 1 ? GRIDN - 1 : ix);
  iy = iy < 0 ? 0 : (iy > GRIDN - 1 ? GRIDN - 1 : iy);
  float cx = PC0 + ((float)ix + 0.5f) * 0.8f;
  float cy = PC0 + ((float)iy + 0.5f) * 0.8f;
  float f[8] = {x, y, zc, x - cx, y - cy, zc, ptf[idx * 2], ptf[idx * 2 + 1]};
  size_t flat = (size_t)(b * GRIDN + iy) * GRIDN + ix;
  maskp[flat] = 1.0f;
  int* gp = (int*)(grid + flat * CH);
  for (int c = 0; c < CH; ++c) {
    float s = 0.0f;
#pragma unroll
    for (int k = 0; k < 8; ++k) s = fmaf(f[k], sw[k * CH + c], s);
    s = fmaxf(s * g[c] + bi[c], 0.0f);
    atomicMax(gp + c, __float_as_int(s));
  }
}

// ---------------------------------------------------------------------------
// 3x3 conv 256->256 (padding 1, optional stride 2) via WMMA implicit GEMM.
// One wave: 16 output x's (M) x all 256 output channels (16 n-tiles).
// 4 waves/block share the weight stream through a double-buffered LDS stage
// filled by async global->LDS copies (ASYNCcnt), issued one k-step ahead.
// Epilogue fuses BN affine / ReLU / occupancy mask / residual add.
// ---------------------------------------------------------------------------
template <bool MASKF, bool RELU_PRE, bool ADD_RES, bool RELU_POST>
__global__ __launch_bounds__(128) void conv3x3_wmma(
    const unsigned short* __restrict__ in, const unsigned short* __restrict__ wp,
    const float* __restrict__ gamma, const float* __restrict__ beta,
    const float* __restrict__ maskp, const unsigned short* __restrict__ res,
    unsigned short* __restrict__ out,
    int Hin, int Win, int Hout, int Wout, int stride) {
  __shared__ unsigned short sB[2][16 * 512];          // 2 x 16 KB B-fragment stage

  const int lane   = threadIdx.x & 31;
  const int tilesX = (Wout + 15) >> 4;
  const int nTiles = NB * Hout * tilesX;
  int tileRaw = blockIdx.x * 4 + (threadIdx.x >> 5);  // wave-uniform
  const bool active = tileRaw < nTiles;
  int tile = active ? tileRaw : (nTiles - 1);         // clamp: keep all waves in barriers
  int tx = tile % tilesX;
  int t2 = tile / tilesX;
  int yo = t2 % Hout;
  int b  = t2 / Hout;
  int x0 = tx << 4;
  const int r   = lane & 15;
  const int hi  = lane >> 4;
  const int xol = x0 + r;                             // this lane's A-matrix row pixel

  // async copy of one k-step's 16 KB of packed B fragments into LDS buffer bi
  const unsigned ldsBase = (unsigned)(size_t)(&sB[0][0]);
  auto issueB = [&](int s, int bi) {
    const char* src = (const char*)wp + (size_t)s * STEP_BYTES + (size_t)threadIdx.x * 16;
    unsigned dst = ldsBase + (unsigned)bi * STEP_BYTES + (unsigned)threadIdx.x * 16u;
#pragma unroll
    for (int k = 0; k < 8; ++k)
      asm volatile("global_load_async_to_lds_b128 %0, %1, off"
                   :: "v"(dst + (unsigned)(k * 2048)),
                      "v"((unsigned long long)(size_t)(src + k * 2048))
                   : "memory");
  };

  // A fragment (contiguous k order, same as packed B): lane<16 holds K={0..15},
  // lane>=16 holds K={16..31}; zeros outside the padded image.
  auto loadA = [&](int s) -> v16u {
    int tap = s >> 3, kc = s & 7;
    int dy = tap / 3, dx = tap - dy * 3;
    int yi = yo * stride + dy - 1;
    int xi = xol * stride + dx - 1;
    bool ok = ((unsigned)yi < (unsigned)Hin) && ((unsigned)xi < (unsigned)Win);
    v8u alo = {0, 0, 0, 0, 0, 0, 0, 0};
    v8u ahi = {0, 0, 0, 0, 0, 0, 0, 0};
    if (ok) {
      const unsigned short* ap =
          in + ((size_t)(b * Hin + yi) * Win + xi) * CH + hi * 16 + kc * 32;
      alo = *(const v8u*)ap;            // K = hi*16 + 0..7
      ahi = *(const v8u*)(ap + 8);      // K = hi*16 + 8..15
    }
    v16u Au;
#pragma unroll
    for (int e = 0; e < 8; ++e) { Au[e] = alo[e]; Au[e + 8] = ahi[e]; }
    return Au;
  };

  v8f acc[16];
  const v8f vz = {0.f, 0.f, 0.f, 0.f, 0.f, 0.f, 0.f, 0.f};
#pragma unroll
  for (int i = 0; i < 16; ++i) acc[i] = vz;

  // software pipeline: A one step ahead in regs, B one step ahead in LDS
  v16u Acur = loadA(0);
  issueB(0, 0);
  for (int s = 0; s < KSTEPS; ++s) {
    asm volatile("s_wait_asynccnt 0x0" ::: "memory");   // my quarter of step s landed
    __syncthreads();                                    // everyone's landed; buf[(s+1)&1] free
    if (s + 1 < KSTEPS) issueB(s + 1, (s + 1) & 1);
    v16u Anext = (s + 1 < KSTEPS) ? loadA(s + 1) : Acur;
    const v16bf A = __builtin_bit_cast(v16bf, Acur);
    const unsigned short* bbase = &sB[s & 1][lane * 16];
    // rotate B through registers with distance-2 prefetch: the ds_load for
    // fragment nt+2 is in flight while WMMA nt executes
    v16u Bc = *(const v16u*)(bbase);
    v16u Bn = *(const v16u*)(bbase + 512);
#pragma unroll
    for (int nt = 0; nt < 16; ++nt) {
      v16u Bp = (nt + 2 < 16) ? *(const v16u*)(bbase + (nt + 2) * 512) : Bc;
      acc[nt] = __builtin_amdgcn_wmma_f32_16x16x32_bf16(
          false, A, false, __builtin_bit_cast(v16bf, Bc),
          (short)0, acc[nt], false, false);
      Bc = Bn;
      Bn = Bp;
    }
    Acur = Anext;
  }

  if (!active) return;

  // C/D layout: lane holds channel column n=lane&15, rows m = hi*8 + j.
  float mk[8];
#pragma unroll
  for (int j = 0; j < 8; ++j) {
    int xo2 = x0 + hi * 8 + j;
    float m = 1.0f;
    if (MASKF) m = (xo2 < Wout) ? maskp[(size_t)(b * Hout + yo) * Wout + xo2] : 0.0f;
    mk[j] = m;
  }
#pragma unroll
  for (int nt = 0; nt < 16; ++nt) {
    int c = nt * 16 + r;
    float gc = gamma[c], bc = beta[c];
#pragma unroll
    for (int j = 0; j < 8; ++j) {
      int xo2 = x0 + hi * 8 + j;
      if (xo2 < Wout) {
        float v = acc[nt][j] * gc + bc;
        if (RELU_PRE) v = fmaxf(v, 0.0f);
        if (MASKF) v *= mk[j];
        size_t oidx = ((size_t)(b * Hout + yo) * Wout + xo2) * CH + c;
        if (ADD_RES) v += b2f(res[oidx]);
        if (RELU_POST) v = fmaxf(v, 0.0f);
        out[oidx] = f2b(v);
      }
    }
  }
}

// ---------------------------------------------------------------------------
// Host-side orchestration
// ---------------------------------------------------------------------------
extern "C" void kernel_launch(void* const* d_in, const int* in_sizes, int n_in,
                              void* d_out, int out_size, void* d_ws, size_t ws_size,
                              hipStream_t stream) {
  const float* xyz  = (const float*)d_in[0];
  const float* ptf  = (const float*)d_in[1];
  const int*   bcnt = (const int*)d_in[2];
  const float* mw   = (const float*)d_in[3];
  const float* mg   = (const float*)d_in[4];
  const float* mb   = (const float*)d_in[5];
  const float* w4   = (const float*)d_in[6];
  const float* g4   = (const float*)d_in[7];
  const float* b4   = (const float*)d_in[8];
  const float* w5   = (const float*)d_in[9];
  const float* g5   = (const float*)d_in[10];
  const float* b5   = (const float*)d_in[11];
  float* out = (float*)d_out;
  char*  ws  = (char*)d_ws;

  const size_t gridElems  = (size_t)NB * GRIDN * GRIDN * CH;     // 18,096,128
  const size_t maskElems  = (size_t)NB * GRIDN * GRIDN;          // 70,688
  const size_t act4Bytes  = gridElems * 2;                       // bf16
  const size_t grid5Elems = (size_t)NB * GRIDN5 * GRIDN5 * CH;   // 4,524,032
  const int    WPC        = 9 * CH * CH;                         // packed shorts / conv

  size_t off = 0;
  float*          grid_f = (float*)(ws + off);          off += gridElems * 4;
  float*          maskp  = (float*)(ws + off);          off += maskElems * 4;
  unsigned short* bufX   = (unsigned short*)(ws + off); off += act4Bytes;
  unsigned short* buf1   = (unsigned short*)(ws + off); off += act4Bytes;
  unsigned short* buf2   = (unsigned short*)(ws + off); off += act4Bytes;
  unsigned short* b5a    = (unsigned short*)(ws + off); off += grid5Elems * 2;
  unsigned short* b5b    = (unsigned short*)(ws + off); off += grid5Elems * 2;
  unsigned short* b5c    = (unsigned short*)(ws + off); off += grid5Elems * 2;
  unsigned short* wp4    = (unsigned short*)(ws + off); off += (size_t)5 * WPC * 2;
  unsigned short* wp5    = (unsigned short*)(ws + off); off += (size_t)5 * WPC * 2;
  // total ~220 MB of workspace

  // 1) zero grid + mask (contiguous; d_ws is poisoned each run)
  {
    size_t n = gridElems + maskElems;
    zero_f32<<<(unsigned)((n + 255) / 256), 256, 0, stream>>>(grid_f, n);
  }
  // 2) pre-pack all 10 conv weight tensors into WMMA B-fragment order
  for (int i = 0; i < 5; ++i) {
    pack_weights<<<(9 * 8 * 16 * 32 + 127) / 128, 128, 0, stream>>>(
        w4 + (size_t)i * WPC, wp4 + (size_t)i * WPC);
    pack_weights<<<(9 * 8 * 16 * 32 + 127) / 128, 128, 0, stream>>>(
        w5 + (size_t)i * WPC, wp5 + (size_t)i * WPC);
  }
  // 3) points: featurize + MLP + scatter-max
  int npts = in_sizes[0] / 3;
  points_kernel<<<(npts + 255) / 256, 256, 0, stream>>>(xyz, ptf, bcnt, mw, mg, mb,
                                                        grid_f, maskp, npts);
  // 4) grid f32 -> bf16 activations
  cvt_f32_bf16<<<(unsigned)((gridElems + 255) / 256), 256, 0, stream>>>(grid_f, bufX, gridElems);

  const int blocks4 = (NB * GRIDN * ((GRIDN + 15) / 16) + 3) / 4;    // 1128
  const int blocks5 = (NB * GRIDN5 * ((GRIDN5 + 15) / 16) + 3) / 4;  // 282

  // 5) conv4 chain (Sparse2DBasicBlockV + Sparse2DBasicBlock)
  conv3x3_wmma<true, true, false, false><<<blocks4, 128, 0, stream>>>(
      bufX, wp4 + 0 * WPC, g4 + 0 * CH, b4 + 0 * CH, maskp, nullptr, buf1,
      GRIDN, GRIDN, GRIDN, GRIDN, 1);                                 // x0
  conv3x3_wmma<true, true, false, false><<<blocks4, 128, 0, stream>>>(
      buf1, wp4 + 1 * WPC, g4 + 1 * CH, b4 + 1 * CH, maskp, nullptr, buf2,
      GRIDN, GRIDN, GRIDN, GRIDN, 1);
  conv3x3_wmma<true, false, true, true><<<blocks4, 128, 0, stream>>>(
      buf2, wp4 + 2 * WPC, g4 + 2 * CH, b4 + 2 * CH, maskp, buf1, bufX,
      GRIDN, GRIDN, GRIDN, GRIDN, 1);                                 // x1
  conv3x3_wmma<true, true, false, false><<<blocks4, 128, 0, stream>>>(
      bufX, wp4 + 3 * WPC, g4 + 3 * CH, b4 + 3 * CH, maskp, nullptr, buf1,
      GRIDN, GRIDN, GRIDN, GRIDN, 1);
  conv3x3_wmma<true, false, true, true><<<blocks4, 128, 0, stream>>>(
      buf1, wp4 + 4 * WPC, g4 + 4 * CH, b4 + 4 * CH, maskp, bufX, buf2,
      GRIDN, GRIDN, GRIDN, GRIDN, 1);                                 // x_conv4

  // 6) conv5: stride-2 down + two Dense2DBasicBlocks
  conv3x3_wmma<false, true, false, false><<<blocks5, 128, 0, stream>>>(
      buf2, wp5 + 0 * WPC, g5 + 0 * CH, b5 + 0 * CH, nullptr, nullptr, b5a,
      GRIDN, GRIDN, GRIDN5, GRIDN5, 2);                               // y0
  conv3x3_wmma<false, true, false, false><<<blocks5, 128, 0, stream>>>(
      b5a, wp5 + 1 * WPC, g5 + 1 * CH, b5 + 1 * CH, nullptr, nullptr, b5b,
      GRIDN5, GRIDN5, GRIDN5, GRIDN5, 1);
  conv3x3_wmma<false, false, true, true><<<blocks5, 128, 0, stream>>>(
      b5b, wp5 + 2 * WPC, g5 + 2 * CH, b5 + 2 * CH, nullptr, b5a, b5c,
      GRIDN5, GRIDN5, GRIDN5, GRIDN5, 1);                             // y1
  conv3x3_wmma<false, true, false, false><<<blocks5, 128, 0, stream>>>(
      b5c, wp5 + 3 * WPC, g5 + 3 * CH, b5 + 3 * CH, nullptr, nullptr, b5b,
      GRIDN5, GRIDN5, GRIDN5, GRIDN5, 1);
  conv3x3_wmma<false, false, true, true><<<blocks5, 128, 0, stream>>>(
      b5b, wp5 + 4 * WPC, g5 + 4 * CH, b5 + 4 * CH, nullptr, b5c, b5a,
      GRIDN5, GRIDN5, GRIDN5, GRIDN5, 1);                             // y

  // 7) outputs: (x_conv4 NCHW fp32, y NCHW fp32) concatenated
  {
    size_t n1 = (size_t)NB * CH * GRIDN * GRIDN;
    nhwc_to_nchw<<<(unsigned)((n1 + 255) / 256), 256, 0, stream>>>(buf2, out, GRIDN, GRIDN);
    size_t n2 = (size_t)NB * CH * GRIDN5 * GRIDN5;
    nhwc_to_nchw<<<(unsigned)((n2 + 255) / 256), 256, 0, stream>>>(b5a, out + n1, GRIDN5, GRIDN5);
  }
}
